// WindowAttention_89137751261676
// MI455X (gfx1250) — matrix operand order repaired and checked by reading, not compiled, not run on hardware
//
#include <hip/hip_runtime.h>
#include <hip/hip_bf16.h>

typedef __attribute__((ext_vector_type(16))) _Float16 v16h;
typedef __attribute__((ext_vector_type(8)))  _Float16 v8h;
typedef __attribute__((ext_vector_type(4)))  _Float16 v4h;
typedef __attribute__((ext_vector_type(8)))  float    v8f;

#define WMMA(A,B,C) __builtin_amdgcn_wmma_f32_16x16x32_f16(false,(A),false,(B),(short)0,(C),false,false)

static __device__ __forceinline__ v16h cat8(v8h lo, v8h hi) {
  return __builtin_shufflevector(lo, hi, 0,1,2,3,4,5,6,7,8,9,10,11,12,13,14,15);
}

// ---------------- constants ----------------
// B=64, H=W=64, C=256, HEADS=8, HEAD_DIM=32, WS=8, S=64, nW=64, DISP=4
#define SCALE   0.17677669529663687f
#define QKVW_H  196608L          // 768*256
#define PROJW_H 65536L           // 256*256
#define BUF_H   67108864L        // 64*8*64*64*32 halves per q/k/vT/attn buffer

#define USE_ASYNC_LDS 1

// ---------------- weight f32 -> f16 ----------------
__global__ __launch_bounds__(256)
void wcvt_kernel(const float* __restrict__ qkv_w, const float* __restrict__ proj_w,
                 _Float16* __restrict__ dst) {
  int idx = blockIdx.x * 256 + threadIdx.x;             // 262144 total
  float v = (idx < (int)QKVW_H) ? qkv_w[idx] : proj_w[idx - (int)QKVW_H];
  dst[idx] = (_Float16)v;
}

// ---------------- fused roll + QKV GEMM + window scatter ----------------
// One 384-thread block (12 waves) per 16-row M-tile. The rolled x-tile is
// converted to f16 once into LDS; each wave owns 64 of the 768 N columns.
__global__ __launch_bounds__(384)
void qkv_gemm_kernel(const float* __restrict__ x, const _Float16* __restrict__ w,
                     _Float16* __restrict__ q, _Float16* __restrict__ k,
                     _Float16* __restrict__ vt) {
  __shared__ _Float16 ldsA[16 * 256];                   // 8 KB
  const int tid = threadIdx.x;
  const long m0 = (long)blockIdx.x * 16;

  // cooperative stage: 16 rows x 256 K, f32 -> f16, cyclic roll folded in
  for (int idx = tid; idx < 1024; idx += 384) {
    const int row = idx >> 6;                           // 0..15
    const int c4  = (idx & 63) << 2;                    // K offset, step 4
    const long m = m0 + row;
    const int b  = (int)(m >> 12);
    const int ii = (int)((m >> 6) & 63);
    const int jj = (int)(m & 63);
    const int isrc = (ii + 4) & 63, jsrc = (jj + 4) & 63;
    const float4 f = *(const float4*)(x + (((long)b * 64 + isrc) * 64 + jsrc) * 256 + c4);
    v4h h; h[0] = (_Float16)f.x; h[1] = (_Float16)f.y;
           h[2] = (_Float16)f.z; h[3] = (_Float16)f.w;
    *(v4h*)&ldsA[row * 256 + c4] = h;
  }
  __syncthreads();

  const int wave = tid >> 5;                            // 0..11 == nt
  const int lane = tid & 31;
  const int l16 = lane & 15, hi = lane >> 4;
  const int ko = hi * 16;
  const int nt = wave;

  v8f acc[4] = {v8f{0.f}, v8f{0.f}, v8f{0.f}, v8f{0.f}};
#pragma unroll
  for (int K0 = 0; K0 < 256; K0 += 32) {
    const _Float16* ar = &ldsA[l16 * 256 + K0 + ko];
    const v16h a = cat8(*(const v8h*)ar, *(const v8h*)(ar + 8));
#pragma unroll
    for (int t = 0; t < 4; ++t) {
      const int n = nt * 64 + t * 16 + l16;
      const v16h bb = *(const v16h*)(w + (long)n * 256 + K0 + hi * 16);
      acc[t] = WMMA(a, bb, acc[t]);
    }
  }

  // Scatter into windowed layouts (f16). q gets SCALE folded in; v stored transposed.
#pragma unroll
  for (int t = 0; t < 4; ++t) {
    const int n    = nt * 64 + t * 16 + l16;
    const int part = n >> 8;                            // 0=q 1=k 2=v
    const int nn   = n & 255;
    const int head = nn >> 5, d = nn & 31;
#pragma unroll
    for (int r = 0; r < 8; ++r) {
      const long mr = m0 + r + 8 * hi;
      const int b2 = (int)(mr >> 12);
      const int i2 = (int)((mr >> 6) & 63);
      const int j2 = (int)(mr & 63);
      const int win = ((i2 >> 3) << 3) | (j2 >> 3);
      const int s   = ((i2 & 7) << 3) | (j2 & 7);
      const long base = (((long)b2 * 8 + head) * 64 + win) * 2048;
      const float v = acc[t][r];
      if (part == 0)      q [base + (long)s * 32 + d] = (_Float16)(v * SCALE);
      else if (part == 1) k [base + (long)s * 32 + d] = (_Float16)v;
      else                vt[base + (long)d * 64 + s] = (_Float16)v;
    }
  }
}

// ---------------- attention: one block per (b, head, window) ----------------
__global__ __launch_bounds__(128)
void attn_kernel(const _Float16* __restrict__ q, const _Float16* __restrict__ k,
                 const _Float16* __restrict__ vt, const float* __restrict__ pos,
                 _Float16* __restrict__ attn) {
  __shared__ float    lpos[225];            // pos_emb [15x15]
  __shared__ _Float16 probs[64 * 64];       // per-block softmax probabilities

  const int tid = threadIdx.x;
  if (tid < 225) lpos[tid] = pos[tid];
  __syncthreads();

  const int wave = tid >> 5, lane = tid & 31;
  const int l16 = lane & 15, hi = lane >> 4;
  const long bw = blockIdx.x;               // ((b*8+head)*64 + win)
  const int  win = (int)(bw & 63);
  const long hwbase = bw * 2048;
  const int  m0 = wave * 16;                // 16 query rows per wave
  const int  ko = hi * 16;

  // --- scores = q @ k^T (4 x 16x16 tiles, single K=32 step) ---
  const _Float16* qr = q + hwbase + (long)(m0 + l16) * 32;
  const v16h aq = cat8(*(const v8h*)(qr + ko), *(const v8h*)(qr + ko + 8));

  v8f sc[4];
#pragma unroll
  for (int t = 0; t < 4; ++t) {
    const int j = t * 16 + l16;
    const v16h bk = *(const v16h*)(k + hwbase + (long)j * 32 + hi * 16);
    v8f z = {0.f};
    sc[t] = WMMA(aq, bk, z);
  }

  // --- relative-position bias + analytic shift mask ---
  const int wrow = win >> 3, wcol = win & 7;
#pragma unroll
  for (int t = 0; t < 4; ++t) {
    const int j = t * 16 + l16;
#pragma unroll
    for (int r = 0; r < 8; ++r) {
      const int i = m0 + r + 8 * hi;
      const int ri = (i >> 3) - (j >> 3) + 7;
      const int ci = (i & 7) - (j & 7) + 7;
      float v = sc[t][r] + lpos[ri * 15 + ci];
      if (wrow == 7 && ((i >= 32) != (j >= 32)))           v = -1e30f;
      if (wcol == 7 && (((i & 7) >= 4) != ((j & 7) >= 4))) v = -1e30f;
      sc[t][r] = v;
    }
  }

  // --- softmax over 64 keys: rows live in 16-lane halves ---
  float rcp[8];
#pragma unroll
  for (int r = 0; r < 8; ++r) {
    float mx = fmaxf(fmaxf(sc[0][r], sc[1][r]), fmaxf(sc[2][r], sc[3][r]));
    for (int off = 1; off < 16; off <<= 1) mx = fmaxf(mx, __shfl_xor(mx, off, 16));
    float sum = 0.f;
#pragma unroll
    for (int t = 0; t < 4; ++t) { sc[t][r] = __expf(sc[t][r] - mx); sum += sc[t][r]; }
    for (int off = 1; off < 16; off <<= 1) sum += __shfl_xor(sum, off, 16);
    rcp[r] = __frcp_rn(sum);
  }

  // --- probs (C layout) -> LDS row-major, re-read as A fragments ---
#pragma unroll
  for (int t = 0; t < 4; ++t) {
    const int j = t * 16 + l16;
#pragma unroll
    for (int r = 0; r < 8; ++r) {
      const int i = m0 + r + 8 * hi;
      probs[i * 64 + j] = (_Float16)sc[t][r];
    }
  }
  __syncthreads();

  // --- out = probs @ v : K=64 in 2 steps, N=32 in 2 tiles ---
  v8f o[2] = {v8f{0.f}, v8f{0.f}};
#pragma unroll
  for (int kt = 0; kt < 2; ++kt) {
    const _Float16* pr = &probs[(m0 + l16) * 64 + kt * 32];
    const v16h ap = cat8(*(const v8h*)(pr + ko), *(const v8h*)(pr + ko + 8));
#pragma unroll
    for (int t = 0; t < 2; ++t) {
      const int d = t * 16 + l16;
      const v16h bv = *(const v16h*)(vt + hwbase + (long)d * 64 + kt * 32 + hi * 16);
      o[t] = WMMA(ap, bv, o[t]);
    }
  }

  // --- normalize by rowsum, store f16 [.., s, d] ---
#pragma unroll
  for (int t = 0; t < 2; ++t) {
    const int d = t * 16 + l16;
#pragma unroll
    for (int r = 0; r < 8; ++r) {
      const int s = m0 + r + 8 * hi;
      attn[hwbase + (long)s * 32 + d] = (_Float16)(o[t][r] * rcp[r]);
    }
  }
}

// ---------------- un-window + output proj + bias + reverse roll ----------------
// 256-thread block = 2 M-tiles x 4 N-wavetiles. A tiles (un-windowed gather)
// are staged to LDS with async global->LDS copies (ASYNCcnt path).
__global__ __launch_bounds__(256)
void proj_gemm_kernel(const _Float16* __restrict__ attn, const _Float16* __restrict__ w,
                      const float* __restrict__ bias, float* __restrict__ out) {
  __shared__ _Float16 ldsA[2][16 * 256];                // 16 KB
  const int tid = threadIdx.x;
  const long m0b = (long)blockIdx.x * 32;

  // stage 1024 chunks of 8 halves (16 B) : windowed attn -> row-major LDS
  for (int c = tid; c < 1024; c += 256) {
    const int tileM = c >> 9;
    const int row   = (c >> 5) & 15;
    const int ch    = c & 31;                           // 8 heads x 4 d-chunks
    const int head  = ch >> 2;
    const int d0    = (ch & 3) << 3;
    const long m = m0b + tileM * 16 + row;
    const int b  = (int)(m >> 12);
    const int ii = (int)((m >> 6) & 63);
    const int jj = (int)(m & 63);
    const int win = ((ii >> 3) << 3) | (jj >> 3);
    const int s   = ((ii & 7) << 3) | (jj & 7);
    const _Float16* src = attn + (((long)b * 8 + head) * 64 + win) * 2048 + (long)s * 32 + d0;
    _Float16* dst = &ldsA[tileM][row * 256 + ch * 8];
#if USE_ASYNC_LDS
    const unsigned lds_off = (unsigned)(unsigned long long)dst;   // LDS aperture: addr[31:0]
    const unsigned long long ga = (unsigned long long)src;
    asm volatile("global_load_async_to_lds_b128 %0, %1, off"
                 :: "v"(lds_off), "v"(ga) : "memory");
#else
    *(v8h*)dst = *(const v8h*)src;
#endif
  }
#if USE_ASYNC_LDS
  asm volatile("s_wait_asynccnt 0" ::: "memory");
#endif
  __syncthreads();

  const int wave = tid >> 5;
  const int lane = tid & 31;
  const int l16 = lane & 15, hi = lane >> 4;
  const int mtl = wave >> 2;                            // 0..1 local M tile
  const int nt  = wave & 3;                             // 4 x 64 = N 256
  const int ko = hi * 16;
  const long m0 = m0b + mtl * 16;

  v8f acc[4] = {v8f{0.f}, v8f{0.f}, v8f{0.f}, v8f{0.f}};
#pragma unroll
  for (int K0 = 0; K0 < 256; K0 += 32) {
    const _Float16* ar = &ldsA[mtl][l16 * 256 + K0 + ko];
    const v16h a = cat8(*(const v8h*)ar, *(const v8h*)(ar + 8));
#pragma unroll
    for (int t = 0; t < 4; ++t) {
      const int n = nt * 64 + t * 16 + l16;
      const v16h bb = *(const v16h*)(w + (long)n * 256 + K0 + hi * 16);
      acc[t] = WMMA(a, bb, acc[t]);
    }
  }

#pragma unroll
  for (int t = 0; t < 4; ++t) {
    const int n = nt * 64 + t * 16 + l16;
    const float bn = bias[n];
#pragma unroll
    for (int r = 0; r < 8; ++r) {
      const long mr = m0 + r + 8 * hi;
      const int b2 = (int)(mr >> 12);
      const int i2 = (int)((mr >> 6) & 63);
      const int j2 = (int)(mr & 63);
      const int id = (i2 + 4) & 63, jd = (j2 + 4) & 63;   // reverse cyclic shift
      out[(((long)b2 * 64 + id) * 64 + jd) * 256 + n] = acc[t][r] + bn;
    }
  }
}

extern "C" void kernel_launch(void* const* d_in, const int* in_sizes, int n_in,
                              void* d_out, int out_size, void* d_ws, size_t ws_size,
                              hipStream_t stream) {
  const float* x      = (const float*)d_in[0];
  const float* qkv_w  = (const float*)d_in[1];
  const float* proj_w = (const float*)d_in[2];
  const float* proj_b = (const float*)d_in[3];
  const float* pos    = (const float*)d_in[4];
  float* out = (float*)d_out;

  _Float16* ws   = (_Float16*)d_ws;
  _Float16* qw_h = ws;                       // 196608
  _Float16* pw_h = ws + QKVW_H;              // 65536
  _Float16* qb   = ws + QKVW_H + PROJW_H;
  _Float16* kb   = qb + BUF_H;
  _Float16* vtb  = kb + BUF_H;
  _Float16* ab   = vtb + BUF_H;

  wcvt_kernel<<<1024, 256, 0, stream>>>(qkv_w, proj_w, qw_h);
  qkv_gemm_kernel<<<16384, 384, 0, stream>>>(x, qw_h, qb, kb, vtb);
  attn_kernel<<<32768, 128, 0, stream>>>(qb, kb, vtb, pos, ab);
  proj_gemm_kernel<<<8192, 256, 0, stream>>>(ab, pw_h, proj_b, out);
}
